// Model_14903536517401
// MI455X (gfx1250) — matrix operand order repaired and checked by reading, not compiled, not run on hardware
//
#include <hip/hip_runtime.h>
#include <cstdint>

// Problem constants (from reference)
#define BB    64
#define TT    512
#define EE    128
#define HH    512
#define G4    2048   // 4*HH
#define PREDN 96

typedef unsigned short u16;
typedef __attribute__((ext_vector_type(16))) __bf16 v16bf;
typedef __attribute__((ext_vector_type(8)))  float  v8f;

union B16Frag { v16bf v; uint4 q[2]; };

__device__ __forceinline__ u16 f2bf(float f) {
    union { float f; unsigned u; } x; x.f = f;
    unsigned u = x.u + 0x7FFFu + ((x.u >> 16) & 1u);   // RNE
    return (u16)(u >> 16);
}

// A fragment: 16x32 bf16, row-major source with `stride` (elements).
// ISA layout: lanes 0-15 -> rows 0-15 K{0..7,16..23}; lanes 16-31 -> rows 0-15 K{8..15,24..31}
__device__ __forceinline__ void load_fragA(B16Frag& fr, const u16* __restrict__ src,
                                           int stride, int k0, int lane) {
    const int r = lane & 15, half = lane >> 4;
    const u16* p = src + r * stride + k0 + half * 8;
    fr.q[0] = *(const uint4*)(p);
    fr.q[1] = *(const uint4*)(p + 16);
}

// B fragment: 32x16 bf16 where B[k][n] = W[n][k], W row-major with row length `stride`.
// lanes 0-15 -> col n, K{0..7,16..23} ; lanes 16-31 -> col n, K{8..15,24..31} (same as A)
__device__ __forceinline__ void load_fragB(B16Frag& fr, const u16* __restrict__ W,
                                           int stride, int nbase, int k0, int lane) {
    const int col = lane & 15, kb = (lane >> 4) * 8;
    const u16* p = W + (nbase + col) * stride + k0 + kb;
    fr.q[0] = *(const uint4*)(p); fr.q[1] = *(const uint4*)(p + 16);
}

__device__ __forceinline__ float sigm(float x) { return 1.0f / (1.0f + __expf(-x)); }

// ---------------- prep kernels ----------------
__global__ void k_f32_to_bf16(const float* __restrict__ in, u16* __restrict__ out, int n) {
    int i = blockIdx.x * blockDim.x + threadIdx.x;
    if (i < n) out[i] = f2bf(in[i]);
}
// dec_in0 = x_dec[:, -1, :] -> bf16 (rows x cols, strided rows)
__global__ void k_slice_to_bf16(const float* __restrict__ in, int row_stride,
                                int rows, int cols, u16* __restrict__ out) {
    int i = blockIdx.x * blockDim.x + threadIdx.x;
    if (i < rows * cols) {
        int r = i / cols, c = i % cols;
        out[i] = f2bf(in[r * row_stride + c]);
    }
}
__global__ void k_bias_combine(const float* __restrict__ a, const float* __restrict__ b,
                               float* __restrict__ out, int n) {
    int i = blockIdx.x * blockDim.x + threadIdx.x;
    if (i < n) out[i] = a[i] + b[i];
}

// ---------------- fused LSTM cell (one timestep, one layer) ----------------
// z = x @ W_ih^T + h_prev @ W_hh^T + bias ; gates -> c_out, h_out(bf16)
// grid: (32 jtiles, 4 mtiles), block: 128 threads = 4 waves (one wave per gate)
__global__ void __launch_bounds__(128)
k_lstm_cell(const u16* __restrict__ x, int x_stride, int Kx,
            const u16* __restrict__ W_ih,          // (G4 x Kx) bf16
            const u16* __restrict__ h_prev,        // (BB x HH) bf16
            const u16* __restrict__ W_hh,          // (G4 x HH) bf16
            const float* __restrict__ bias,        // (G4) f32 (b_ih+b_hh)
            const float* __restrict__ c_prev,      // (BB x HH) f32
            float* __restrict__ c_out,             // (BB x HH) f32
            u16* __restrict__ h_out)               // (BB x HH) bf16
{
    __shared__ float zsh[4][16][16];
    const int lane = threadIdx.x & 31;
    const int w    = threadIdx.x >> 5;   // gate id: 0=i 1=f 2=g 3=o
    const int jt   = blockIdx.x;         // hidden col tile 0..31
    const int mt   = blockIdx.y;         // batch tile 0..3
    const int nbase = w * HH + jt * 16;  // row block in W / col block in z

    v8f acc = {};
    const u16* xrow = x + mt * 16 * x_stride;
    for (int k0 = 0; k0 < Kx; k0 += 32) {
        B16Frag a, b;
        load_fragA(a, xrow, x_stride, k0, lane);
        load_fragB(b, W_ih, Kx, nbase, k0, lane);
        acc = __builtin_amdgcn_wmma_f32_16x16x32_bf16(false, a.v, false, b.v,
                                                      (short)0, acc, false, false);
    }
    const u16* hrow = h_prev + mt * 16 * HH;
    for (int k0 = 0; k0 < HH; k0 += 32) {
        B16Frag a, b;
        load_fragA(a, hrow, HH, k0, lane);
        load_fragB(b, W_hh, HH, nbase, k0, lane);
        acc = __builtin_amdgcn_wmma_f32_16x16x32_bf16(false, a.v, false, b.v,
                                                      (short)0, acc, false, false);
    }

    // C/D layout: VGPR r -> M = r (lanes 0-15) / 8+r (lanes 16-31), N = lane&15
    const int col = lane & 15, rbase = (lane < 16) ? 0 : 8;
    #pragma unroll
    for (int r = 0; r < 8; ++r) zsh[w][rbase + r][col] = acc[r];
    __syncthreads();

    for (int e = threadIdx.x; e < 256; e += 128) {
        const int m = e >> 4, j = e & 15;
        const int J = jt * 16 + j;       // hidden unit
        const int M = mt * 16 + m;       // batch row
        const float zi = zsh[0][m][j] + bias[J];
        const float zf = zsh[1][m][j] + bias[HH + J];
        const float zg = zsh[2][m][j] + bias[2 * HH + J];
        const float zo = zsh[3][m][j] + bias[3 * HH + J];
        const float cn = sigm(zf) * c_prev[M * HH + J] + sigm(zi) * tanhf(zg);
        const float hn = sigm(zo) * tanhf(cn);
        c_out[M * HH + J] = cn;
        h_out[M * HH + J] = f2bf(hn);
    }
}

// ---------------- FC head: out = h1 @ W_fc^T + b_fc ----------------
// grid: 8 ntiles, block 128 = 4 waves (one per mtile). Writes f32 to d_out and bf16 next input.
__global__ void __launch_bounds__(128)
k_fc(const u16* __restrict__ h1,           // (BB x HH) bf16
     const u16* __restrict__ Wfc,          // (EE x HH) bf16
     const float* __restrict__ bfc,        // (EE) f32
     float* __restrict__ out, int step_off,// out[b, step, :] ; step_off = step*EE
     u16* __restrict__ x_next)             // (BB x EE) bf16
{
    const int lane = threadIdx.x & 31;
    const int mt   = threadIdx.x >> 5;     // 0..3
    const int nt   = blockIdx.x;           // 0..7
    v8f acc = {};
    const u16* hrow = h1 + mt * 16 * HH;
    for (int k0 = 0; k0 < HH; k0 += 32) {
        B16Frag a, b;
        load_fragA(a, hrow, HH, k0, lane);
        load_fragB(b, Wfc, HH, nt * 16, k0, lane);
        acc = __builtin_amdgcn_wmma_f32_16x16x32_bf16(false, a.v, false, b.v,
                                                      (short)0, acc, false, false);
    }
    const int col = lane & 15, rbase = (lane < 16) ? 0 : 8;
    #pragma unroll
    for (int r = 0; r < 8; ++r) {
        const int m = mt * 16 + rbase + r;
        const int n = nt * 16 + col;
        const float v = acc[r] + bfc[n];
        out[m * (PREDN * EE) + step_off + n] = v;
        x_next[m * EE + n] = f2bf(v);
    }
}

// ---------------- host ----------------
static inline char* carve(char*& p, size_t bytes) {
    uintptr_t u = ((uintptr_t)p + 255) & ~(uintptr_t)255;
    char* r = (char*)u;
    p = r + bytes;
    return r;
}

extern "C" void kernel_launch(void* const* d_in, const int* in_sizes, int n_in,
                              void* d_out, int out_size, void* d_ws, size_t ws_size,
                              hipStream_t stream) {
    (void)in_sizes; (void)n_in; (void)out_size; (void)ws_size;
    const float* x_enc = (const float*)d_in[0];
    const float* x_dec = (const float*)d_in[2];
    const float* W_ih0 = (const float*)d_in[4];
    const float* W_hh0 = (const float*)d_in[5];
    const float* b_ih0 = (const float*)d_in[6];
    const float* b_hh0 = (const float*)d_in[7];
    const float* W_ih1 = (const float*)d_in[8];
    const float* W_hh1 = (const float*)d_in[9];
    const float* b_ih1 = (const float*)d_in[10];
    const float* b_hh1 = (const float*)d_in[11];
    const float* W_fc  = (const float*)d_in[12];
    const float* b_fc  = (const float*)d_in[13];
    float* out = (float*)d_out;

    char* p = (char*)d_ws;
    u16*   xenc_bf = (u16*)  carve(p, (size_t)BB * TT * EE * sizeof(u16));   // 8.4 MB
    u16*   Y0      = (u16*)  carve(p, (size_t)TT * BB * HH * sizeof(u16));   // 33.6 MB (layer-0 h history)
    u16*   Wih0b   = (u16*)  carve(p, (size_t)G4 * EE * sizeof(u16));
    u16*   Whh0b   = (u16*)  carve(p, (size_t)G4 * HH * sizeof(u16));
    u16*   Wih1b   = (u16*)  carve(p, (size_t)G4 * HH * sizeof(u16));
    u16*   Whh1b   = (u16*)  carve(p, (size_t)G4 * HH * sizeof(u16));
    u16*   Wfcb    = (u16*)  carve(p, (size_t)EE * HH * sizeof(u16));
    float* bias0   = (float*)carve(p, G4 * sizeof(float));
    float* bias1   = (float*)carve(p, G4 * sizeof(float));
    u16*   hzero   = (u16*)  carve(p, (size_t)BB * HH * sizeof(u16));
    float* c0a     = (float*)carve(p, (size_t)BB * HH * sizeof(float));
    float* c0b     = (float*)carve(p, (size_t)BB * HH * sizeof(float));
    u16*   h1a     = (u16*)  carve(p, (size_t)BB * HH * sizeof(u16));
    u16*   h1b     = (u16*)  carve(p, (size_t)BB * HH * sizeof(u16));
    float* c1a     = (float*)carve(p, (size_t)BB * HH * sizeof(float));
    float* c1b     = (float*)carve(p, (size_t)BB * HH * sizeof(float));
    u16*   h0a     = (u16*)  carve(p, (size_t)BB * HH * sizeof(u16));
    u16*   h0b     = (u16*)  carve(p, (size_t)BB * HH * sizeof(u16));
    u16*   xd0     = (u16*)  carve(p, (size_t)BB * EE * sizeof(u16));
    u16*   xd1     = (u16*)  carve(p, (size_t)BB * EE * sizeof(u16));

    // ---- prep: bf16 weight/input conversion, bias fusion, zero state ----
    {
        int n;
        n = BB * TT * EE; k_f32_to_bf16<<<(n + 255) / 256, 256, 0, stream>>>(x_enc, xenc_bf, n);
        n = G4 * EE;      k_f32_to_bf16<<<(n + 255) / 256, 256, 0, stream>>>(W_ih0, Wih0b, n);
        n = G4 * HH;      k_f32_to_bf16<<<(n + 255) / 256, 256, 0, stream>>>(W_hh0, Whh0b, n);
        n = G4 * HH;      k_f32_to_bf16<<<(n + 255) / 256, 256, 0, stream>>>(W_ih1, Wih1b, n);
        n = G4 * HH;      k_f32_to_bf16<<<(n + 255) / 256, 256, 0, stream>>>(W_hh1, Whh1b, n);
        n = EE * HH;      k_f32_to_bf16<<<(n + 255) / 256, 256, 0, stream>>>(W_fc, Wfcb, n);
        k_bias_combine<<<(G4 + 255) / 256, 256, 0, stream>>>(b_ih0, b_hh0, bias0, G4);
        k_bias_combine<<<(G4 + 255) / 256, 256, 0, stream>>>(b_ih1, b_hh1, bias1, G4);
        // dec_in0 = x_dec[:, LABEL+PRED-1, :]
        k_slice_to_bf16<<<(BB * EE + 255) / 256, 256, 0, stream>>>(
            x_dec + (size_t)143 * EE, 144 * EE, BB, EE, xd0);
        hipMemsetAsync(hzero, 0, (size_t)BB * HH * sizeof(u16), stream);
        hipMemsetAsync(c0a,   0, (size_t)BB * HH * sizeof(float), stream);
        hipMemsetAsync(h1a,   0, (size_t)BB * HH * sizeof(u16), stream);
        hipMemsetAsync(c1a,   0, (size_t)BB * HH * sizeof(float), stream);
    }

    const dim3 cgrid(32, 4), cblk(128);

    // ---- encoder layer 0: x_enc (Kx=128), h history stored in Y0 ----
    const u16* hp0 = hzero;
    float *cp0 = c0a, *cn0 = c0b;
    for (int t = 0; t < TT; ++t) {
        u16* hout = Y0 + (size_t)t * BB * HH;
        k_lstm_cell<<<cgrid, cblk, 0, stream>>>(
            xenc_bf + (size_t)t * EE, TT * EE, EE,
            Wih0b, hp0, Whh0b, bias0, cp0, cn0, hout);
        hp0 = hout;
        float* tmp = cp0; cp0 = cn0; cn0 = tmp;
    }
    // final layer-0 state: h in hp0 (= Y0 + 511*...), c in cp0

    // ---- encoder layer 1: input = Y0[t] (Kx=512) ----
    const u16* hp1 = h1a; u16* hn1 = h1b;
    float *cp1 = c1a, *cn1 = c1b;
    for (int t = 0; t < TT; ++t) {
        k_lstm_cell<<<cgrid, cblk, 0, stream>>>(
            Y0 + (size_t)t * BB * HH, HH, HH,
            Wih1b, hp1, Whh1b, bias1, cp1, cn1, hn1);
        const u16* th = hp1; hp1 = hn1; hn1 = (u16*)th;
        float* tc = cp1; cp1 = cn1; cn1 = tc;
    }
    // final layer-1 state: h in hp1, c in cp1

    // ---- decoder: 96 autoregressive steps ----
    const u16* xp = xd0; u16* xn = xd1;
    u16* hn0 = h0a; u16* hn0b = h0b;
    for (int s = 0; s < PREDN; ++s) {
        // layer 0 (Kx = 128)
        k_lstm_cell<<<cgrid, cblk, 0, stream>>>(
            xp, EE, EE, Wih0b, hp0, Whh0b, bias0, cp0, cn0, hn0);
        // layer 1 (input = new h0, Kx = 512)
        k_lstm_cell<<<cgrid, cblk, 0, stream>>>(
            hn0, HH, HH, Wih1b, hp1, Whh1b, bias1, cp1, cn1, hn1);
        // fc head -> d_out[:, s, :] and next decoder input (bf16)
        k_fc<<<dim3(8), cblk, 0, stream>>>(hn1, Wfcb, b_fc, out, s * EE, xn);

        // rotate state buffers
        hp0 = hn0; { u16* t = hn0; hn0 = hn0b; hn0b = t; }
        { float* t = cp0; cp0 = cn0; cn0 = t; }
        { const u16* t = hp1; hp1 = hn1; hn1 = (u16*)t; }
        { float* t = cp1; cp1 = cn1; cn1 = t; }
        { const u16* t = xp; xp = xn; xn = (u16*)t; }
    }
}
